// SABlock_32203664786002
// MI455X (gfx1250) — matrix operand order repaired and checked by reading, not compiled, not run on hardware
//
#include <hip/hip_runtime.h>
#include <hip/hip_bf16.h>

// ---------------------------------------------------------------------------
// Self-attention block for MI455X (gfx1250, wave32, WMMA, async LDS copies).
//   pack:  w_qkv, w_out  f32 -> bf16 (once)
//   qkv  = x @ w_qkv^T                 (GEMM, bf16 WMMA, f32 accum, bf16 out)
//   attn = flash-attention per head    (QK^T + online softmax + PV, WMMA)
//   out  = attn @ w_out^T + b_out      (GEMM, bf16 WMMA, f32 out)
// bf16 operands are staged LDS-side with global_load_async_to_lds_b128.
// ---------------------------------------------------------------------------

typedef __attribute__((ext_vector_type(16))) __bf16 v16bf;
typedef __attribute__((ext_vector_type(8)))  float  v8f;

union V16BF { unsigned int u[8]; v16bf v; };

static __device__ __forceinline__ unsigned int bf16_rne(float f) {
    unsigned int x = __float_as_uint(f);
    return (x + 0x7FFFu + ((x >> 16) & 1u)) >> 16;   // round-to-nearest-even
}
static __device__ __forceinline__ unsigned int pack2_bf16(float lo, float hi) {
    return bf16_rne(lo) | (bf16_rne(hi) << 16);
}
static __device__ __forceinline__ v8f zero_v8() {
    v8f z;
#pragma unroll
    for (int i = 0; i < 8; ++i) z[i] = 0.0f;
    return z;
}
// A-fragment (16x32 bf16) pair index for VGPR i, lane-half (0: lanes 0-15)
static __device__ __forceinline__ int a_pair(int i, int half) {
    return i + half * 8;
}
// async bf16 tile copy: one 16B chunk global -> LDS (ASYNCcnt tracked)
static __device__ __forceinline__ void async_b128(unsigned int* lds_dst,
                                                  const unsigned int* gsrc) {
    unsigned int d = (unsigned int)(size_t)lds_dst;   // low 32b = LDS offset
    asm volatile("global_load_async_to_lds_b128 %0, %1, off"
                 :: "v"(d), "v"(gsrc) : "memory");
}
static __device__ __forceinline__ void wait_async0() {
    asm volatile("s_wait_asynccnt 0x0" ::: "memory");
}

// ---------------------------------------------------------------------------
// pack pairs of f32 -> one u32 of two bf16
// ---------------------------------------------------------------------------
__global__ __launch_bounds__(256)
void sab_pack_bf16(const float* __restrict__ src, unsigned int* __restrict__ dst,
                   int npairs) {
    int i = blockIdx.x * 256 + threadIdx.x;
    if (i < npairs) {
        const float* g = src + 2 * (size_t)i;
        dst[i] = pack2_bf16(g[0], g[1]);
    }
}

// ---------------------------------------------------------------------------
// GEMM:  C[m][n] = sum_k A[m][k] * W[n][k]  (+ bias[n])
// W is pre-packed bf16 pairs [N][K/2].  A is f32 or bf16-pairs per template.
// Block: 128 threads (4 waves).  Tile: 128(M) x 64(N), K-step 32.
// Each wave: 32 rows x 64 cols = 8 WMMAs per K-step (2 A-frags x 4 B-frags).
// ---------------------------------------------------------------------------
#define GS 20   // LDS row stride (dwords): 16 pairs + pad, 16B-aligned rows

template<bool A_BF16, bool OUT_BF16>
__global__ __launch_bounds__(128)
void sab_gemm(const void* __restrict__ Aany,
              const unsigned int* __restrict__ Wb,
              const float* __restrict__ bias,
              void* __restrict__ Cany,
              int M, int N, int K)
{
    __shared__ unsigned int lds_a[128 * GS];
    __shared__ unsigned int lds_b[64 * GS];

    const int t    = threadIdx.x;
    const int lane = t & 31;
    const int wave = t >> 5;
    const int half = lane >> 4;
    const int l16  = lane & 15;
    const int n0   = blockIdx.x * 64;
    const int m0   = blockIdx.y * 128;
    const int Kp   = K >> 1;

    v8f acc[2][4];
#pragma unroll
    for (int ms = 0; ms < 2; ++ms)
#pragma unroll
        for (int nt = 0; nt < 4; ++nt) acc[ms][nt] = zero_v8();

    for (int k0 = 0; k0 < K; k0 += 32) {
        // ---- stage W tile: 64 rows x 4 x B128 async copies ----
#pragma unroll
        for (int i = 0; i < 2; ++i) {
            int u = t + i * 128;                 // 0..255
            int r = u >> 2, c = u & 3;
            async_b128(&lds_b[r * GS + c * 4],
                       Wb + (size_t)(n0 + r) * Kp + (k0 >> 1) + c * 4);
        }
        // ---- stage A tile ----
        if constexpr (A_BF16) {
            const unsigned int* Ab = (const unsigned int*)Aany;
#pragma unroll
            for (int i = 0; i < 4; ++i) {
                int u = t + i * 128;             // 0..511
                int r = u >> 2, c = u & 3;
                async_b128(&lds_a[r * GS + c * 4],
                           Ab + (size_t)(m0 + r) * Kp + (k0 >> 1) + c * 4);
            }
        } else {
            const float* Af = (const float*)Aany;
#pragma unroll
            for (int i = 0; i < 16; ++i) {
                int u = t + i * 128;             // 0..2047
                int r = u >> 4, p = u & 15;
                const float* g = Af + (size_t)(m0 + r) * K + k0 + 2 * p;
                lds_a[r * GS + p] = pack2_bf16(g[0], g[1]);
            }
            if (k0 + 32 < K)                     // global_prefetch_b8
                __builtin_prefetch(Af + (size_t)(m0 + t) * K + k0 + 32, 0, 1);
        }
        wait_async0();
        __syncthreads();

        // ---- fragments + WMMA ----
        V16BF a[2];
#pragma unroll
        for (int ms = 0; ms < 2; ++ms) {
            const int base = (wave * 32 + ms * 16 + l16) * GS;
#pragma unroll
            for (int i = 0; i < 8; ++i) a[ms].u[i] = lds_a[base + a_pair(i, half)];
        }
#pragma unroll
        for (int nt = 0; nt < 4; ++nt) {
            V16BF b;
            const int base = (nt * 16 + l16) * GS + half * 8;
#pragma unroll
            for (int i = 0; i < 8; ++i) b.u[i] = lds_b[base + i];
#pragma unroll
            for (int ms = 0; ms < 2; ++ms)
                acc[ms][nt] = __builtin_amdgcn_wmma_f32_16x16x32_bf16(
                    false, a[ms].v, false, b.v, (short)0, acc[ms][nt], false, false);
        }
        __syncthreads();
    }

    // ---- store: row = r + half*8 per 16x16 tile ----
#pragma unroll
    for (int ms = 0; ms < 2; ++ms) {
        const int rowBase = m0 + wave * 32 + ms * 16 + half * 8;
#pragma unroll
        for (int nt = 0; nt < 4; ++nt) {
            const int col = n0 + nt * 16 + l16;
#pragma unroll
            for (int r = 0; r < 8; ++r) {
                if constexpr (OUT_BF16) {
                    ((unsigned short*)Cany)[(size_t)(rowBase + r) * N + col] =
                        (unsigned short)bf16_rne(acc[ms][nt][r]);
                } else {
                    ((float*)Cany)[(size_t)(rowBase + r) * N + col] =
                        acc[ms][nt][r] + bias[col];
                }
            }
        }
    }
}

// ---------------------------------------------------------------------------
// Flash attention on bf16 qkv:  qkvb is [B*S][1536] u32 (3072 bf16).
// q at u32 col head*32.., k at 512+head*32.., v at 1024+head*32..
// One block = 64 query rows of one (batch, head); 128 threads (4 waves).
// ---------------------------------------------------------------------------
#define AS 36   // LDS row stride (dwords): 32 pairs + pad, 16B-aligned rows

__global__ __launch_bounds__(128)
void sab_attn_flash(const unsigned int* __restrict__ qkvb,
                    unsigned short* __restrict__ attb,
                    int S)
{
    __shared__ unsigned int lds_q[64 * AS];     // [row][dim-pair]
    __shared__ unsigned int lds_k[64 * AS];     // [key][dim-pair]
    __shared__ unsigned int lds_v[64 * AS];     // [dim][key-pair]
    __shared__ unsigned int lds_p[4 * 16 * AS]; // per-wave P 16x64 bf16

    const int t    = threadIdx.x;
    const int lane = t & 31;
    const int wave = t >> 5;
    const int half = lane >> 4;
    const int l16  = lane & 15;

    const int q0    = blockIdx.x * 64;
    const int head  = blockIdx.y;
    const int batch = blockIdx.z;
    const size_t rowbase = (size_t)batch * S;
    const int hc = head * 32;                   // u32 column of this head

    // ---- stage Q via async B128: 64 rows x 8 chunks ----
#pragma unroll
    for (int i = 0; i < 4; ++i) {
        int u = t + i * 128;                    // 0..511
        int r = u >> 3, c = u & 7;
        async_b128(&lds_q[r * AS + c * 4],
                   qkvb + (rowbase + q0 + r) * 1536 + hc + c * 4);
    }
    wait_async0();
    __syncthreads();

    V16BF qa[2];
#pragma unroll
    for (int kt = 0; kt < 2; ++kt) {
        const int base = (wave * 16 + l16) * AS + kt * 16;
#pragma unroll
        for (int i = 0; i < 8; ++i) qa[kt].u[i] = lds_q[base + a_pair(i, half)];
    }
    __syncthreads();

    float mstate[8], lstate[8];
#pragma unroll
    for (int r = 0; r < 8; ++r) { mstate[r] = -1e30f; lstate[r] = 0.0f; }
    v8f o[4];
#pragma unroll
    for (int i = 0; i < 4; ++i) o[i] = zero_v8();

    for (int j0 = 0; j0 < S; j0 += 64) {
        // ---- K: async row-major copy ----
#pragma unroll
        for (int i = 0; i < 4; ++i) {
            int u = t + i * 128;
            int r = u >> 3, c = u & 7;
            async_b128(&lds_k[r * AS + c * 4],
                       qkvb + (rowbase + j0 + r) * 1536 + 512 + hc + c * 4);
        }
        // ---- V: transpose 2x2 bf16 blocks (pairs along keys per dim) ----
#pragma unroll
        for (int i = 0; i < 8; ++i) {
            int u  = t + i * 128;               // 0..1023
            int dp = u >> 5, kp = u & 31;
            unsigned int a = qkvb[(rowbase + j0 + 2 * kp)     * 1536 + 1024 + hc + dp];
            unsigned int b = qkvb[(rowbase + j0 + 2 * kp + 1) * 1536 + 1024 + hc + dp];
            lds_v[(2 * dp)     * AS + kp] = (a & 0xFFFFu) | (b << 16);
            lds_v[(2 * dp + 1) * AS + kp] = (a >> 16) | (b & 0xFFFF0000u);
        }
        wait_async0();
        __syncthreads();

        // ---- S = Q K^T * scale : 16x64 per wave ----
        v8f s[4];
#pragma unroll
        for (int nt = 0; nt < 4; ++nt) {
            s[nt] = zero_v8();
#pragma unroll
            for (int kt = 0; kt < 2; ++kt) {
                V16BF b;
                const int base = (nt * 16 + l16) * AS + kt * 16 + half * 8;
#pragma unroll
                for (int i = 0; i < 8; ++i) b.u[i] = lds_k[base + i];
                s[nt] = __builtin_amdgcn_wmma_f32_16x16x32_bf16(
                    false, qa[kt].v, false, b.v, (short)0, s[nt], false, false);
            }
            s[nt] *= 0.125f;                    // DIM_HEAD^-0.5, exact
        }

        // ---- online softmax (row = r + half*8, replicated in 16-lane group) ----
#pragma unroll
        for (int r = 0; r < 8; ++r) {
            float mx = fmaxf(fmaxf(s[0][r], s[1][r]), fmaxf(s[2][r], s[3][r]));
#pragma unroll
            for (int off = 8; off >= 1; off >>= 1)
                mx = fmaxf(mx, __shfl_xor(mx, off, 16));
            const float mnew = fmaxf(mstate[r], mx);
            const float corr = __expf(mstate[r] - mnew);
            float rs = 0.0f;
#pragma unroll
            for (int nt = 0; nt < 4; ++nt) {
                float p = __expf(s[nt][r] - mnew);
                s[nt][r] = p;
                rs += p;
            }
#pragma unroll
            for (int off = 8; off >= 1; off >>= 1)
                rs += __shfl_xor(rs, off, 16);
            lstate[r] = lstate[r] * corr + rs;
            mstate[r] = mnew;
#pragma unroll
            for (int nt = 0; nt < 4; ++nt) o[nt][r] *= corr;
        }

        // ---- P: C-layout -> per-wave LDS -> A-layout ----
        unsigned short* pp = (unsigned short*)lds_p;
#pragma unroll
        for (int nt = 0; nt < 4; ++nt)
#pragma unroll
            for (int r = 0; r < 8; ++r)
                pp[(wave * 16 + r + half * 8) * (2 * AS) + nt * 16 + l16] =
                    (unsigned short)bf16_rne(s[nt][r]);

        V16BF pa[2];
#pragma unroll
        for (int kt = 0; kt < 2; ++kt) {
            const int base = (wave * 16 + l16) * AS + kt * 16;
#pragma unroll
            for (int i = 0; i < 8; ++i) pa[kt].u[i] = lds_p[base + a_pair(i, half)];
        }

        // ---- O += P V ----
#pragma unroll
        for (int nt = 0; nt < 4; ++nt) {
#pragma unroll
            for (int kt = 0; kt < 2; ++kt) {
                V16BF b;
                const int base = (nt * 16 + l16) * AS + kt * 16 + half * 8;
#pragma unroll
                for (int i = 0; i < 8; ++i) b.u[i] = lds_v[base + i];
                o[nt] = __builtin_amdgcn_wmma_f32_16x16x32_bf16(
                    false, pa[kt].v, false, b.v, (short)0, o[nt], false, false);
            }
        }
        __syncthreads();
    }

    // ---- finalize: O /= l, store bf16 [B*S][1024] ----
#pragma unroll
    for (int r = 0; r < 8; ++r) {
        const float inv = 1.0f / lstate[r];
        const size_t row = rowbase + q0 + wave * 16 + half * 8 + r;
#pragma unroll
        for (int nt = 0; nt < 4; ++nt)
            attb[row * 1024 + head * 64 + nt * 16 + l16] =
                (unsigned short)bf16_rne(o[nt][r] * inv);
    }
}

// ---------------------------------------------------------------------------
extern "C" void kernel_launch(void* const* d_in, const int* in_sizes, int n_in,
                              void* d_out, int out_size, void* d_ws, size_t ws_size,
                              hipStream_t stream) {
    (void)n_in; (void)out_size; (void)ws_size;
    const float* x     = (const float*)d_in[0];   // [2,2048,1024]
    const float* w_qkv = (const float*)d_in[1];   // [3072,1024]
    const float* w_out = (const float*)d_in[2];   // [1024,1024]
    const float* b_out = (const float*)d_in[3];   // [1024]
    float* out = (float*)d_out;

    const int BS = in_sizes[0] / 1024;            // 4096 = B*S
    const int S  = BS / 2;                        // 2048

    // workspace (u32 units): wq | wo | qkvb | attb  (~40 MB total)
    unsigned int* wq   = (unsigned int*)d_ws;                 // 3072*512
    unsigned int* wo   = wq + 3072 * 512;                     // 1024*512
    unsigned int* qkvb = wo + 1024 * 512;                     // BS*1536
    unsigned short* attb = (unsigned short*)(qkvb + (size_t)BS * 1536); // BS*1024

    dim3 blk(128);
    // 0) pack weights f32 -> bf16 pairs (once per call; cheap)
    sab_pack_bf16<<<dim3((3072 * 512) / 256), dim3(256), 0, stream>>>(w_qkv, wq, 3072 * 512);
    sab_pack_bf16<<<dim3((1024 * 512) / 256), dim3(256), 0, stream>>>(w_out, wo, 1024 * 512);
    // 1) QKV projection: M=BS, N=3072, K=1024 (f32 A, bf16 out)
    sab_gemm<false, true><<<dim3(3072 / 64, BS / 128), blk, 0, stream>>>(
        x, wq, nullptr, qkvb, BS, 3072, 1024);
    // 2) flash attention per (64 query rows, head, batch)
    sab_attn_flash<<<dim3(S / 64, 16, 2), blk, 0, stream>>>(qkvb, attb, S);
    // 3) output projection + bias: M=BS, N=1024, K=1024 (bf16 A, f32 out)
    sab_gemm<true, false><<<dim3(1024 / 64, BS / 128), blk, 0, stream>>>(
        attb, wo, b_out, out, BS, 1024, 1024);
}
